// MoT_28681791603141
// MI455X (gfx1250) — matrix operand order repaired and checked by reading, not compiled, not run on hardware
//
#include <hip/hip_runtime.h>
#include <math.h>
#include <stdint.h>

// ---- problem constants ----
#define BB 1
#define SV 2048
#define SA 512
#define DD 1024
#define HH 16
#define LL 2
#define FF 4096
#define DH 64
#define SS (SV + SA)   // 2560

typedef __attribute__((ext_vector_type(16))) __bf16 v16bf;
typedef __attribute__((ext_vector_type(8)))  float  v8f;

// ---------------------------------------------------------------------------
// Helper: build a v16bf fragment from two 16B chunks
// ---------------------------------------------------------------------------
union frag_u { uint4 u[2]; v16bf v; };

// ===========================================================================
// LayerNorm + AdaLN modulate:  out = LN(x) * (1 + msc + tsc) + (msh + tsh)
// one block (256 thr) per token row, D = 1024 (4 elems / thread)
// ===========================================================================
__global__ __launch_bounds__(256) void ln_mod_kernel(
    const float* __restrict__ x,
    const float* __restrict__ msh, const float* __restrict__ tsh,
    const float* __restrict__ msc, const float* __restrict__ tsc,
    __bf16* __restrict__ out)
{
    const int row = blockIdx.x, tid = threadIdx.x;
    const float* xr = x + (size_t)row * DD;
    float4 v = *reinterpret_cast<const float4*>(xr + tid * 4);
    float s  = v.x + v.y + v.z + v.w;
    float ss = v.x * v.x + v.y * v.y + v.z * v.z + v.w * v.w;

    __shared__ float2 red[256];
    red[tid] = make_float2(s, ss);
    __syncthreads();
    for (int st = 128; st > 0; st >>= 1) {
        if (tid < st) {
            red[tid].x += red[tid + st].x;
            red[tid].y += red[tid + st].y;
        }
        __syncthreads();
    }
    const float mean = red[0].x * (1.0f / DD);
    const float var  = red[0].y * (1.0f / DD) - mean * mean;
    const float rstd = rsqrtf(var + 1e-6f);

    const int c = tid * 4;
    float e[4] = {v.x, v.y, v.z, v.w};
    __bf16* outr = out + (size_t)row * DD + c;
#pragma unroll
    for (int k = 0; k < 4; ++k) {
        const float sc = 1.0f + msc[c + k] + tsc[c + k];
        const float sh = msh[c + k] + tsh[c + k];
        outr[k] = (__bf16)((e[k] - mean) * rstd * sc + sh);
    }
}

// ===========================================================================
// RMSNorm * g  followed by RoPE; writes bf16 into joint q/k buffer.
// one block per token row; D = 1024 -> thread owns 4 consecutive = 2 pairs
// ===========================================================================
__global__ __launch_bounds__(256) void qk_rope_kernel(
    const float* __restrict__ lin, const float* __restrict__ g,
    const float* __restrict__ freqs, __bf16* __restrict__ out)
{
    const int row = blockIdx.x, tid = threadIdx.x;
    const float* xr = lin + (size_t)row * DD;
    float4 v = *reinterpret_cast<const float4*>(xr + tid * 4);
    float ss = v.x * v.x + v.y * v.y + v.z * v.z + v.w * v.w;

    __shared__ float red[256];
    red[tid] = ss;
    __syncthreads();
    for (int st = 128; st > 0; st >>= 1) {
        if (tid < st) red[tid] += red[tid + st];
        __syncthreads();
    }
    const float rms = rsqrtf(red[0] * (1.0f / DD) + 1e-6f);

    const int c = tid * 4;
    const float y0 = v.x * rms * g[c + 0];
    const float y1 = v.y * rms * g[c + 1];
    const float y2 = v.z * rms * g[c + 2];
    const float y3 = v.w * rms * g[c + 3];

    const float* fr = freqs + (size_t)row * (DH / 2);
    const int p0 = (c & (DH - 1)) >> 1;       // pair index within head
    const float th0 = fr[p0],     cs0 = cosf(th0), sn0 = sinf(th0);
    const float th1 = fr[p0 + 1], cs1 = cosf(th1), sn1 = sinf(th1);

    __bf16* outr = out + (size_t)row * DD + c;
    outr[0] = (__bf16)(y0 * cs0 - y1 * sn0);
    outr[1] = (__bf16)(y0 * sn0 + y1 * cs0);
    outr[2] = (__bf16)(y2 * cs1 - y3 * sn1);
    outr[3] = (__bf16)(y2 * sn1 + y3 * cs1);
}

// ===========================================================================
// residual:  x += (g1[c] + g2[c]) * t   (vectorized by 4)
// ===========================================================================
__global__ __launch_bounds__(256) void residual_kernel(
    float* __restrict__ x, const float* __restrict__ t,
    const float* __restrict__ g1, const float* __restrict__ g2)
{
    const int i = (blockIdx.x * 256 + threadIdx.x) * 4;
    const int c = i & (DD - 1);
    float4 xv = *reinterpret_cast<float4*>(x + i);
    float4 tv = *reinterpret_cast<const float4*>(t + i);
    xv.x += (g1[c + 0] + g2[c + 0]) * tv.x;
    xv.y += (g1[c + 1] + g2[c + 1]) * tv.y;
    xv.z += (g1[c + 2] + g2[c + 2]) * tv.z;
    xv.w += (g1[c + 3] + g2[c + 3]) * tv.w;
    *reinterpret_cast<float4*>(x + i) = xv;
}

// ===========================================================================
// GEMM:  out[M,N] = A_bf16[M,K] @ W_f32[K,N] + bias[N]   (+epilogue)
//   EPI 0 : fp32 row-major store
//   EPI 1 : gelu(tanh) -> bf16 row-major store
//   EPI 3 : bf16 TRANSPOSED store  out[col*ldT + row]   (for V -> [D][S])
// block = 256 thr (8 waves); C-tile = 128x64; wave tile = 16x64 (4 wmma/kstep)
// A tile is staged with GLOBAL_LOAD_ASYNC_TO_LDS_B128 (ASYNCcnt path);
// B tile goes through VGPRs because it converts fp32 -> bf16 in flight.
// ===========================================================================
template <int EPI>
__global__ __launch_bounds__(256) void gemm_kernel(
    const __bf16* __restrict__ A, const float* __restrict__ W,
    const float* __restrict__ bias, void* __restrict__ outp,
    int M, int N, int K, int ldT)
{
    __shared__ __align__(16) __bf16 As[128 * 48];   // padded stride 48
    __shared__ __align__(16) __bf16 Bs[64 * 48];    // stored transposed [n][k]

    const int tid  = threadIdx.x;
    const int m0   = blockIdx.x * 128;
    const int n0   = blockIdx.y * 64;
    const int wv   = tid >> 5;
    const int lane = tid & 31;
    const int r    = lane & 15;
    const int half = lane >> 4;

    v8f acc[4] = {};

    const int ar = tid >> 1, ah = tid & 1;          // A staging: 32B / thread
    const int bn = tid & 63, bk = (tid >> 6) << 3;  // B staging: 8 f32 -> bf16

    // LDS byte offset of this thread's A-stripe (low 32 bits of the generic
    // address are the addrspace(3) offset on gfx1250's aperture scheme)
    const unsigned ldsa =
        (unsigned)(uintptr_t)(&As[ar * 48 + ah * 16]);

    for (int kk = 0; kk < K; kk += 32) {
        { // stage A tile: async global -> LDS, no VGPR round trip.
          // IOFFSET is added to BOTH the LDS and the global address, so the
          // second 16B half shares base registers with offset:16.
            const __bf16* src = A + (size_t)(m0 + ar) * K + kk + ah * 16;
            asm volatile(
                "global_load_async_to_lds_b128 %0, %1, off\n\t"
                "global_load_async_to_lds_b128 %0, %1, off offset:16"
                :: "v"(ldsa), "v"(src) : "memory");
        }
        { // stage B tile transposed with fp32 -> bf16 conversion
            const float* wsrc = W + (size_t)(kk + bk) * N + n0 + bn;
            __align__(16) __bf16 tb[8];
#pragma unroll
            for (int j = 0; j < 8; ++j) tb[j] = (__bf16)wsrc[(size_t)j * N];
            *reinterpret_cast<uint4*>(&Bs[bn * 48 + bk]) =
                *reinterpret_cast<uint4*>(tb);
            if (kk + 32 < K) __builtin_prefetch(wsrc + (size_t)32 * N, 0, 1);
        }
        // wait for this wave's async LDS writes, then workgroup barrier
        asm volatile("s_wait_asynccnt 0x0" ::: "memory");
        __syncthreads();

        // A fragment: row = lane%16; K = half*16..+15
        frag_u af;
        {
            const __bf16* p = &As[(wv * 16 + r) * 48 + half * 16];
            af.u[0] = *reinterpret_cast<const uint4*>(p);
            af.u[1] = *reinterpret_cast<const uint4*>(p + 8);
        }
#pragma unroll
        for (int t4 = 0; t4 < 4; ++t4) {
            // B fragment: col = lane%16; K = half*16..+15 (contiguous in Bs)
            frag_u bf;
            const __bf16* p = &Bs[(t4 * 16 + r) * 48 + half * 16];
            bf.u[0] = *reinterpret_cast<const uint4*>(p);
            bf.u[1] = *reinterpret_cast<const uint4*>(p + 8);
            acc[t4] = __builtin_amdgcn_wmma_f32_16x16x32_bf16(
                false, af.v, false, bf.v, (short)0, acc[t4], false, false);
        }
        __syncthreads();
    }

    // epilogue (C layout: lane r = col, vgpr j = row j + 8*half)
#pragma unroll
    for (int t4 = 0; t4 < 4; ++t4) {
        const int col = n0 + t4 * 16 + r;
        const float b = bias[col];
#pragma unroll
        for (int j = 0; j < 8; ++j) {
            const int row = m0 + wv * 16 + j + half * 8;
            float val = acc[t4][j] + b;
            if (EPI == 0) {
                reinterpret_cast<float*>(outp)[(size_t)row * N + col] = val;
            } else if (EPI == 1) {
                const float g = 0.5f * val *
                    (1.0f + tanhf(0.7978845608f * (val + 0.044715f * val * val * val)));
                reinterpret_cast<__bf16*>(outp)[(size_t)row * N + col] = (__bf16)g;
            } else {
                reinterpret_cast<__bf16*>(outp)[(size_t)col * ldT + row] = (__bf16)val;
            }
        }
    }
}

// ===========================================================================
// Flash attention, one wave per (16-row q tile, head).
// qc,kc : [S][D] bf16 (post rope);  vt : [D][S] bf16 (transposed);  ob : [S][D]
// mask:  allowed(q,k) = (k < SV) || (q >= SV && k <= q)
// ===========================================================================
__global__ __launch_bounds__(256) void attn_kernel(
    const __bf16* __restrict__ qc, const __bf16* __restrict__ kc,
    const __bf16* __restrict__ vt, __bf16* __restrict__ ob)
{
    __shared__ __align__(16) __bf16 plds[8][16 * 32];

    const int tid  = threadIdx.x;
    const int wv   = tid >> 5;
    const int lane = tid & 31;
    const int r    = lane & 15;
    const int half = lane >> 4;

    const int gw = blockIdx.x * 8 + wv;
    const int qt = gw >> 4;       // / H
    const int h  = gw & (HH - 1);
    const int m0 = qt * 16;

    // Q A-fragments (K = DH = 64 -> two k=32 fragments)
    v16bf qf[2];
    const __bf16* qb = qc + (size_t)(m0 + r) * DD + h * DH;
#pragma unroll
    for (int f = 0; f < 2; ++f) {
        frag_u t;
        const __bf16* p = qb + f * 32 + half * 16;
        t.u[0] = *reinterpret_cast<const uint4*>(p);
        t.u[1] = *reinterpret_cast<const uint4*>(p + 8);
        qf[f] = t.v;
    }

    v8f o[4] = {};
    float mrow[8], lrow[8];
#pragma unroll
    for (int j = 0; j < 8; ++j) { mrow[j] = -3.0e38f; lrow[j] = 0.0f; }

    const bool isAct = (m0 >= SV);
    const int  kvmax = isAct ? (m0 + 16) : SV;
    const int  nch   = (kvmax + 31) >> 5;

    for (int ch = 0; ch < nch; ++ch) {
        const int kv0 = ch << 5;

        // scores: 2 subtiles of 16 kv each, 2 wmma (k=32) per subtile
        v8f sf[2];
#pragma unroll
        for (int t2 = 0; t2 < 2; ++t2) {
            v8f sa = {};
            const __bf16* kb = kc + (size_t)(kv0 + t2 * 16 + r) * DD + h * DH;
#pragma unroll
            for (int g2 = 0; g2 < 2; ++g2) {
                frag_u t;
                const __bf16* p = kb + g2 * 32 + half * 16;
                t.u[0] = *reinterpret_cast<const uint4*>(p);
                t.u[1] = *reinterpret_cast<const uint4*>(p + 8);
                sa = __builtin_amdgcn_wmma_f32_16x16x32_bf16(
                    false, qf[g2], false, t.v, (short)0, sa, false, false);
            }
            sf[t2] = sa;
        }

        const bool needMask = isAct && (kv0 >= SV);
#pragma unroll
        for (int t2 = 0; t2 < 2; ++t2) {
#pragma unroll
            for (int j = 0; j < 8; ++j) {
                float s = sf[t2][j] * 0.125f;   // 1/sqrt(64)
                if (needMask) {
                    const int kcol = kv0 + t2 * 16 + r;
                    const int qrow = m0 + j + half * 8;
                    if (kcol > qrow) s = -1.0e9f;
                }
                sf[t2][j] = s;
            }
        }

        // online softmax (row reductions across 16-lane half groups)
#pragma unroll
        for (int j = 0; j < 8; ++j) {
            float v = fmaxf(sf[0][j], sf[1][j]);
            v = fmaxf(v, __shfl_xor(v, 1, 32));
            v = fmaxf(v, __shfl_xor(v, 2, 32));
            v = fmaxf(v, __shfl_xor(v, 4, 32));
            v = fmaxf(v, __shfl_xor(v, 8, 32));
            const float mnew  = fmaxf(mrow[j], v);
            const float alpha = __expf(mrow[j] - mnew);
            const float p0 = __expf(sf[0][j] - mnew);
            const float p1 = __expf(sf[1][j] - mnew);
            float rs = p0 + p1;
            rs += __shfl_xor(rs, 1, 32);
            rs += __shfl_xor(rs, 2, 32);
            rs += __shfl_xor(rs, 4, 32);
            rs += __shfl_xor(rs, 8, 32);
            lrow[j] = lrow[j] * alpha + rs;
            mrow[j] = mnew;
#pragma unroll
            for (int t3 = 0; t3 < 4; ++t3) o[t3][j] *= alpha;
            const int prow = j + half * 8;
            plds[wv][prow * 32 + r]      = (__bf16)p0;
            plds[wv][prow * 32 + 16 + r] = (__bf16)p1;
        }
        __builtin_amdgcn_wave_barrier();   // keep LDS store->load in order

        // P as A-fragment (16x32)
        frag_u pf;
        {
            const __bf16* p = &plds[wv][r * 32 + half * 16];
            pf.u[0] = *reinterpret_cast<const uint4*>(p);
            pf.u[1] = *reinterpret_cast<const uint4*>(p + 8);
        }
        // P @ V : 4 dh tiles of 16; V fragments contiguous thanks to [D][S]
#pragma unroll
        for (int t3 = 0; t3 < 4; ++t3) {
            frag_u vf;
            const __bf16* p =
                vt + (size_t)(h * DH + t3 * 16 + r) * SS + kv0 + half * 16;
            vf.u[0] = *reinterpret_cast<const uint4*>(p);
            vf.u[1] = *reinterpret_cast<const uint4*>(p + 8);
            o[t3] = __builtin_amdgcn_wmma_f32_16x16x32_bf16(
                false, pf.v, false, vf.v, (short)0, o[t3], false, false);
        }
    }

    // normalize and store (row-major bf16)
#pragma unroll
    for (int j = 0; j < 8; ++j) {
        const float inv = 1.0f / lrow[j];
        const int row = m0 + j + half * 8;
#pragma unroll
        for (int t3 = 0; t3 < 4; ++t3) {
            ob[(size_t)row * DD + h * DH + t3 * 16 + r] = (__bf16)(o[t3][j] * inv);
        }
    }
}

// ===========================================================================
// host orchestration
// ===========================================================================
extern "C" void kernel_launch(void* const* d_in, const int* in_sizes, int n_in,
                              void* d_out, int out_size, void* d_ws, size_t ws_size,
                              hipStream_t stream)
{
    (void)in_sizes; (void)n_in; (void)out_size; (void)ws_size;

    const float* vid = (const float*)d_in[0];
    const float* act = (const float*)d_in[1];
    const float* vfr = (const float*)d_in[2];
    const float* afr = (const float*)d_in[3];
    const float* vtm = (const float*)d_in[4];
    const float* atm = (const float*)d_in[5];
    const float* Wq  = (const float*)d_in[6];
    const float* Wk  = (const float*)d_in[7];
    const float* Wv  = (const float*)d_in[8];
    const float* Wo  = (const float*)d_in[9];
    const float* bq  = (const float*)d_in[10];
    const float* bk  = (const float*)d_in[11];
    const float* bv  = (const float*)d_in[12];
    const float* bo  = (const float*)d_in[13];
    const float* gq  = (const float*)d_in[14];
    const float* gk  = (const float*)d_in[15];
    const float* mod = (const float*)d_in[16];
    const float* W1  = (const float*)d_in[17];
    const float* b1  = (const float*)d_in[18];
    const float* W2  = (const float*)d_in[19];
    const float* b2  = (const float*)d_in[20];
    // d_in[21] = attention mask: structure known analytically, not needed

    float* x = (float*)d_out;   // resident activations; output = concat(x0,x1)

    // workspace carve-up (~55 MB)
    char* wsp = (char*)d_ws;
    __bf16* abuf = (__bf16*)wsp; wsp += (size_t)SS * DD * 2;
    float*  tmp  = (float*) wsp; wsp += (size_t)SS * DD * 4;
    __bf16* qcb  = (__bf16*)wsp; wsp += (size_t)SS * DD * 2;
    __bf16* kcb  = (__bf16*)wsp; wsp += (size_t)SS * DD * 2;
    __bf16* vtb  = (__bf16*)wsp; wsp += (size_t)SS * DD * 2;   // [D][S]
    __bf16* obb  = (__bf16*)wsp; wsp += (size_t)SS * DD * 2;
    __bf16* h1b  = (__bf16*)wsp; wsp += (size_t)SS * FF * 2;

    hipMemcpyAsync(x, vid, (size_t)SV * DD * sizeof(float),
                   hipMemcpyDeviceToDevice, stream);
    hipMemcpyAsync(x + (size_t)SV * DD, act, (size_t)SA * DD * sizeof(float),
                   hipMemcpyDeviceToDevice, stream);

    for (int l = 0; l < LL; ++l) {
        // ---- pre-attention: modulated LN, QKV projections, rope ----
        for (int e = 0; e < 2; ++e) {
            const int    Se  = e ? SA : SV;
            const int    off = e ? SV : 0;
            float*       xe  = x + (size_t)off * DD;
            const float* tm  = e ? atm : vtm;
            const float* fr  = e ? afr : vfr;
            const size_t wi  = (size_t)(e * LL + l);
            const float* mrow = mod + wi * 6 * DD;

            ln_mod_kernel<<<Se, 256, 0, stream>>>(
                xe, mrow + 0 * DD, tm + 0 * DD, mrow + 1 * DD, tm + 1 * DD,
                abuf + (size_t)off * DD);

            dim3 g(Se / 128, DD / 64);
            // Q
            gemm_kernel<0><<<g, 256, 0, stream>>>(
                abuf + (size_t)off * DD, Wq + wi * DD * DD, bq + wi * DD,
                tmp, Se, DD, DD, 0);
            qk_rope_kernel<<<Se, 256, 0, stream>>>(
                tmp, gq + wi * DD, fr, qcb + (size_t)off * DD);
            // K
            gemm_kernel<0><<<g, 256, 0, stream>>>(
                abuf + (size_t)off * DD, Wk + wi * DD * DD, bk + wi * DD,
                tmp, Se, DD, DD, 0);
            qk_rope_kernel<<<Se, 256, 0, stream>>>(
                tmp, gk + wi * DD, fr, kcb + (size_t)off * DD);
            // V -> transposed bf16 [D][S]
            gemm_kernel<3><<<g, 256, 0, stream>>>(
                abuf + (size_t)off * DD, Wv + wi * DD * DD, bv + wi * DD,
                vtb + off, Se, DD, DD, SS);
        }

        // ---- joint attention ----
        attn_kernel<<<(SS / 16) * HH / 8, 256, 0, stream>>>(qcb, kcb, vtb, obb);

        // ---- post-attention: O-proj + residual, MLP + residual ----
        for (int e = 0; e < 2; ++e) {
            const int    Se  = e ? SA : SV;
            const int    off = e ? SV : 0;
            float*       xe  = x + (size_t)off * DD;
            const float* tm  = e ? atm : vtm;
            const size_t wi  = (size_t)(e * LL + l);
            const float* mrow = mod + wi * 6 * DD;

            dim3 g(Se / 128, DD / 64);
            gemm_kernel<0><<<g, 256, 0, stream>>>(
                obb + (size_t)off * DD, Wo + wi * DD * DD, bo + wi * DD,
                tmp, Se, DD, DD, 0);
            residual_kernel<<<Se * DD / 1024, 256, 0, stream>>>(
                xe, tmp, mrow + 2 * DD, tm + 2 * DD);

            ln_mod_kernel<<<Se, 256, 0, stream>>>(
                xe, mrow + 3 * DD, tm + 3 * DD, mrow + 4 * DD, tm + 4 * DD,
                abuf + (size_t)off * DD);

            dim3 g1(Se / 128, FF / 64);
            gemm_kernel<1><<<g1, 256, 0, stream>>>(
                abuf + (size_t)off * DD, W1 + wi * DD * FF, b1 + wi * FF,
                h1b, Se, FF, DD, 0);
            dim3 g2(Se / 128, DD / 64);
            gemm_kernel<0><<<g2, 256, 0, stream>>>(
                h1b, W2 + wi * FF * DD, b2 + wi * DD, tmp, Se, DD, FF, 0);
            residual_kernel<<<Se * DD / 1024, 256, 0, stream>>>(
                xe, tmp, mrow + 5 * DD, tm + 5 * DD);
        }
    }
}
